// GSP_37890201485791
// MI455X (gfx1250) — matrix operand order repaired and checked by reading, not compiled, not run on hardware
//
#include <hip/hip_runtime.h>
#include <cstdint>
#include <cstddef>

#define BB   128
#define CC   2048
#define NN   196
#define DD   512
#define KP   64
#define NCLS 100
#define MU_   0.3f
#define EPS_  0.1f
#define STEPS_ 100

typedef __bf16 v16bf __attribute__((ext_vector_type(16)));
typedef __bf16 v8bf  __attribute__((ext_vector_type(8)));
typedef __bf16 v4bf  __attribute__((ext_vector_type(4)));
typedef float  v8f   __attribute__((ext_vector_type(8)));

static __device__ __forceinline__ float lae(float a, float b) {
  // logaddexp
  float mx = fmaxf(a, b), mn = fminf(a, b);
  return mx + log1pf(expf(mn - mx));
}

// Async global->LDS copy of 16 bytes per lane (gfx1250 ASYNCcnt path).
// VDST holds the wave-relative LDS byte offset = low 32 bits of the generic
// shared-memory pointer (LDS aperture keeps the offset in addr[31:0]).
static __device__ __forceinline__ void async_to_lds_b128(void* lds_dst, const void* gsrc) {
  unsigned lds_off = (unsigned)(uintptr_t)lds_dst;
  asm volatile("global_load_async_to_lds_b128 %0, %1, off"
               :: "v"(lds_off), "v"(gsrc)
               : "memory");
}
static __device__ __forceinline__ void wait_async() {
  asm volatile("s_wait_asynccnt 0x0" ::: "memory");
}

// ---------------------------------------------------------------------------
// Kernel 0: embed_w fp32 -> bf16 (one shot; lets the GEMM weight tiles use
// the async-to-LDS path with no conversion in the hot loop)
// ---------------------------------------------------------------------------
__global__ void __launch_bounds__(256) ew_to_bf16(const float* __restrict__ ew,
                                                  __bf16* __restrict__ ewb) {
  size_t i = ((size_t)blockIdx.x * 256 + threadIdx.x) * 4;
  float4 v = *(const float4*)(ew + i);
  v4bf p;
  p[0] = (__bf16)v.x; p[1] = (__bf16)v.y; p[2] = (__bf16)v.z; p[3] = (__bf16)v.w;
  *(v4bf*)(ewb + i) = p;
}

// ---------------------------------------------------------------------------
// Kernel 1: bowT[b,n,d] = bf16( sum_c feat[b,c,n] * ew[d,c] + eb[d] )
// Per batch: (512x2048)@(2048x196). Block tile 64(d) x 128(n), 8 waves.
// A tile async-copied (bf16 weights); B tile fp32->bf16 via 4x4 micro-block
// transpose (4x float4 loads -> 4x ds_store_b64), branch-free clamped
// addressing; padding columns hold garbage and are never stored.
// Output is transposed + bf16: one packed v8bf (16B) store per tile per lane.
// ---------------------------------------------------------------------------
__global__ void __launch_bounds__(256) bow_gemm(const float* __restrict__ feat,
                                                const __bf16* __restrict__ ewb,
                                                const float* __restrict__ eb,
                                                __bf16* __restrict__ bowT) {
  __shared__ __align__(16) __bf16 As[64][32];    // [d_local][k]
  __shared__ __align__(16) __bf16 Bs[128][32];   // [n_local][k]
  const int b  = blockIdx.z;
  const int d0 = blockIdx.y * 64;
  const int n0 = blockIdx.x * 128;
  const int tid  = threadIdx.x;
  const int lane = tid & 31, wave = tid >> 5;
  const int wm = wave & 3, wn = wave >> 2;
  const int half = lane >> 4, l16 = lane & 15;

  v8f acc[4];
#pragma unroll
  for (int t = 0; t < 4; ++t)
#pragma unroll
    for (int r = 0; r < 8; ++r) acc[t][r] = 0.0f;

  const float* featB = feat + (size_t)b * CC * NN;
  const int arow = tid >> 2;           // 0..63
  const int acol = (tid & 3) * 8;      // 0,8,16,24
  const int bk = (tid >> 5) * 4;       // k group 0,4,..,28
  const int bn = (tid & 31) * 4;       // n group 0,4,..,124
  int gnb = n0 + bn;
  gnb = gnb > (NN - 4) ? (NN - 4) : gnb;   // clamp (garbage cols discarded later)

  for (int c0 = 0; c0 < CC; c0 += 32) {
    // A tile: async copy ewb[d0+row][c0+col..col+7] (16B per thread, 4KB total)
    async_to_lds_b128(&As[arow][acol], ewb + (size_t)(d0 + arow) * CC + c0 + acol);
    // B tile: 4x4 micro-block transpose feat[b][c0+bk..+3][gnb..+3] -> Bs
    {
      const float* s0 = featB + (size_t)(c0 + bk) * NN + gnb;
      float4 r0 = *(const float4*)(s0);
      float4 r1 = *(const float4*)(s0 + NN);
      float4 r2 = *(const float4*)(s0 + 2 * NN);
      float4 r3 = *(const float4*)(s0 + 3 * NN);
      float rr[4][4] = {{r0.x, r0.y, r0.z, r0.w},
                        {r1.x, r1.y, r1.z, r1.w},
                        {r2.x, r2.y, r2.z, r2.w},
                        {r3.x, r3.y, r3.z, r3.w}};
#pragma unroll
      for (int j = 0; j < 4; ++j) {
        v4bf p;
        p[0] = (__bf16)rr[0][j]; p[1] = (__bf16)rr[1][j];
        p[2] = (__bf16)rr[2][j]; p[3] = (__bf16)rr[3][j];
        *(v4bf*)&Bs[bn + j][bk] = p;
      }
    }
    wait_async();
    __syncthreads();
    // A fragment: lane holds M=l16, K = half*16+0..15
    v16bf af;
    {
      const __bf16* ap  = &As[16 * wm + l16][half * 16];
      const __bf16* ap2 = &As[16 * wm + l16][half * 16 + 8];
#pragma unroll
      for (int i = 0; i < 8; ++i) { af[i] = ap[i]; af[8 + i] = ap2[i]; }
    }
#pragma unroll
    for (int t = 0; t < 4; ++t) {
      v16bf bf;
      const __bf16* bp = &Bs[64 * wn + 16 * t + l16][half * 16];
#pragma unroll
      for (int i = 0; i < 16; ++i) bf[i] = bp[i];
      acc[t] = __builtin_amdgcn_wmma_f32_16x16x32_bf16(false, af, false, bf,
                                                       (short)0, acc[t], false, false);
    }
    __syncthreads();
  }
  // Epilogue: lane's 8 rows are consecutive d -> single 16B packed store
  const int dbase = d0 + 16 * wm + 8 * half;
#pragma unroll
  for (int t = 0; t < 4; ++t) {
    int n = n0 + 64 * wn + 16 * t + l16;
    if (n >= NN) continue;
    v8bf p;
#pragma unroll
    for (int r = 0; r < 8; ++r) p[r] = (__bf16)(acc[t][r] + eb[dbase + r]);
    *(v8bf*)(bowT + ((size_t)b * NN + n) * DD + dbase) = p;
  }
}

// ---------------------------------------------------------------------------
// Kernel 2: normalized prototypes in bf16
// ---------------------------------------------------------------------------
__global__ void __launch_bounds__(32) proto_norm(const float* __restrict__ p,
                                                 __bf16* __restrict__ pn) {
  int k = blockIdx.x, lane = threadIdx.x;
  float s = 0.0f;
  for (int d = lane; d < DD; d += 32) { float v = p[k * DD + d]; s += v * v; }
  for (int o = 16; o; o >>= 1) s += __shfl_xor(s, o, 32);
  float inv = 1.0f / fmaxf(sqrtf(s), 1e-12f);
  for (int d = lane; d < DD; d += 32) pn[k * DD + d] = (__bf16)(p[k * DD + d] * inv);
}

// ---------------------------------------------------------------------------
// Kernel 3: nrm[b,n] = max(||bowT[b,n,:]||, 1e-12)   (wave per (b,n))
// ---------------------------------------------------------------------------
__global__ void __launch_bounds__(32) bow_norm(const __bf16* __restrict__ bowT,
                                               float* __restrict__ nrm) {
  const int n = blockIdx.x, b = blockIdx.y, lane = threadIdx.x;
  const __bf16* src = bowT + ((size_t)b * NN + n) * DD;
  float s = 0.0f;
#pragma unroll
  for (int i = 0; i < 2; ++i) {
    v8bf v = *(const v8bf*)(src + (lane + 32 * i) * 8);
#pragma unroll
    for (int j = 0; j < 8; ++j) { float f = (float)v[j]; s += f * f; }
  }
  for (int o = 16; o; o >>= 1) s += __shfl_xor(s, o, 32);
  if (lane == 0) nrm[b * NN + n] = fmaxf(sqrtf(s), 1e-12f);
}

// ---------------------------------------------------------------------------
// Kernel 4: logK[b,k,n] = -max(2 - 2*(pn.bow)/nrm, 0)/EPS
// Per batch: (64x512)@(512x196). Both A and B tiles async-copied (bf16);
// the inner loop has no VALU conversion at all.
// ---------------------------------------------------------------------------
__global__ void __launch_bounds__(256) cost_gemm(const __bf16* __restrict__ bowT,
                                                 const __bf16* __restrict__ pn,
                                                 const float* __restrict__ nrm,
                                                 float* __restrict__ logK) {
  __shared__ __align__(16) __bf16 As[64][32];    // [kproto][k]
  __shared__ __align__(16) __bf16 Bs[128][32];   // [n_local][k]
  const int b  = blockIdx.z;
  const int n0 = blockIdx.x * 128;
  const int tid  = threadIdx.x;
  const int lane = tid & 31, wave = tid >> 5;
  const int wm = wave & 3, wn = wave >> 2;
  const int half = lane >> 4, l16 = lane & 15;

  v8f acc[4];
#pragma unroll
  for (int t = 0; t < 4; ++t)
#pragma unroll
    for (int r = 0; r < 8; ++r) acc[t][r] = 0.0f;

  const __bf16* bowB = bowT + (size_t)b * NN * DD;
  const int arow = tid >> 2;
  const int acol = (tid & 3) * 8;

  for (int c0 = 0; c0 < DD; c0 += 32) {
    // A tile: 64x32 bf16 = 4KB, 1 transfer per thread
    async_to_lds_b128(&As[arow][acol], pn + (size_t)arow * DD + c0 + acol);
    // B tile: 128x32 bf16 = 8KB, 2 transfers per thread; rows of bowT are
    // d-contiguous which is exactly the [n][k] LDS layout
#pragma unroll
    for (int i = 0; i < 2; ++i) {
      int lin = tid + 256 * i;         // 0..511
      int row = lin >> 2;              // 0..127
      int q   = (lin & 3) * 8;         // 0,8,16,24
      int gn  = n0 + row;
      gn = gn > (NN - 1) ? (NN - 1) : gn;   // clamp (garbage cols discarded)
      async_to_lds_b128(&Bs[row][q], bowB + (size_t)gn * DD + c0 + q);
    }
    wait_async();
    __syncthreads();
    v16bf af;
    {
      const __bf16* ap  = &As[16 * wm + l16][half * 16];
      const __bf16* ap2 = &As[16 * wm + l16][half * 16 + 8];
#pragma unroll
      for (int i = 0; i < 8; ++i) { af[i] = ap[i]; af[8 + i] = ap2[i]; }
    }
#pragma unroll
    for (int t = 0; t < 4; ++t) {
      v16bf bf;
      const __bf16* bp = &Bs[64 * wn + 16 * t + l16][half * 16];
#pragma unroll
      for (int i = 0; i < 16; ++i) bf[i] = bp[i];
      acc[t] = __builtin_amdgcn_wmma_f32_16x16x32_bf16(false, af, false, bf,
                                                       (short)0, acc[t], false, false);
    }
    __syncthreads();
  }
#pragma unroll
  for (int t = 0; t < 4; ++t) {
    int n = n0 + 64 * wn + 16 * t + l16;
    if (n >= NN) continue;
    float nv = nrm[b * NN + n];
#pragma unroll
    for (int r = 0; r < 8; ++r) {
      int kproto = 16 * wm + r + 8 * half;
      float cost = fmaxf(2.0f - 2.0f * acc[t][r] / nv, 0.0f);
      logK[((size_t)b * KP + kproto) * NN + n] = -cost * (1.0f / EPS_);
    }
  }
}

// ---------------------------------------------------------------------------
// Kernel 5: Sinkhorn dual iteration, one wave per batch (shuffle reductions,
// no barriers in the 100-step serial loop). Emits mixing weights and x_attr.
// ---------------------------------------------------------------------------
__global__ void __launch_bounds__(32) sinkhorn(const float* __restrict__ logK,
                                               float* __restrict__ mw,
                                               float* __restrict__ attr_out) {
  const int b = blockIdx.x, lane = threadIdx.x;
  const float* lk = logK + (size_t)b * KP * NN;
  float rv[7];
  bool valid[7];
#pragma unroll
  for (int j = 0; j < 7; ++j) {
    int n = lane + 32 * j;
    valid[j] = (n < NN);
    float r = -1e30f;
    if (valid[j]) {
      float m = -1e30f;
      for (int k = 0; k < KP; ++k) m = fmaxf(m, lk[(size_t)k * NN + n]);
      float s = 0.0f;
      for (int k = 0; k < KP; ++k) s += expf(lk[(size_t)k * NN + n] - m);
      r = m + logf(s);
    }
    rv[j] = r;
  }
  const float log_col  = logf(1.0f / (float)NN);
  const float log_row0 = logf(1.0f - MU_);
  float v = 0.0f;
  for (int it = 0; it < STEPS_; ++it) {
    float tv[7];
    float mx = -1e30f;
#pragma unroll
    for (int j = 0; j < 7; ++j) {
      float t = valid[j] ? (log_col - lae(v, rv[j])) : -1e30f;
      tv[j] = t;
      mx = fmaxf(mx, t);
    }
    for (int o = 16; o; o >>= 1) mx = fmaxf(mx, __shfl_xor(mx, o, 32));
    float s = 0.0f;
#pragma unroll
    for (int j = 0; j < 7; ++j) if (valid[j]) s += expf(tv[j] - mx);
    for (int o = 16; o; o >>= 1) s += __shfl_xor(s, o, 32);
    v = log_row0 - (mx + logf(s));
  }
  float uv[7];
#pragma unroll
  for (int j = 0; j < 7; ++j) {
    uv[j] = -1e30f;
    if (!valid[j]) continue;
    int n = lane + 32 * j;
    float u = log_col - lae(v, rv[j]);
    uv[j] = u;
    float P0 = expf(u + v);
    mw[b * NN + n] = (1.0f - (float)NN * P0) * (1.0f / MU_);
  }
  for (int k = 0; k < KP; ++k) {
    float s = 0.0f;
#pragma unroll
    for (int j = 0; j < 7; ++j) {
      if (valid[j]) {
        int n = lane + 32 * j;
        s += expf(uv[j] + lk[(size_t)k * NN + n]);
      }
    }
    for (int o = 16; o; o >>= 1) s += __shfl_xor(s, o, 32);
    if (lane == 0) attr_out[b * KP + k] = s * (1.0f / MU_);
  }
}

// ---------------------------------------------------------------------------
// Kernel 6: x_gmp[b,c] = max_n mw[b,n] * feat[b,c,n]  (wave per (b,c))
// ---------------------------------------------------------------------------
__global__ void __launch_bounds__(256) gmp_pool(const float* __restrict__ feat,
                                                const float* __restrict__ mw,
                                                float* __restrict__ xgmp) {
  const int b = blockIdx.y;
  const int c = blockIdx.x * 8 + (threadIdx.x >> 5);
  const int lane = threadIdx.x & 31;
  const float* f = feat + ((size_t)b * CC + c) * NN;
  const float* w = mw + b * NN;
  float m = -1e30f;
  for (int n = lane; n < NN; n += 32) m = fmaxf(m, w[n] * f[n]);
  for (int o = 16; o; o >>= 1) m = fmaxf(m, __shfl_xor(m, o, 32));
  if (lane == 0) xgmp[(size_t)b * CC + c] = m;
}

// ---------------------------------------------------------------------------
// Kernel 7: x_emb_un[b,d] = sum_c x_gmp[b,c]*ew[d,c] + eb[d]
// (128x2048)@(2048x512). Block tile 128(b) x 64(d); B tile async-copied.
// ---------------------------------------------------------------------------
__global__ void __launch_bounds__(256) emb_gemm(const float* __restrict__ xgmp,
                                                const __bf16* __restrict__ ewb,
                                                const float* __restrict__ eb,
                                                float* __restrict__ xembun) {
  __shared__ __align__(16) __bf16 As[128][32];   // [b_local][k]
  __shared__ __align__(16) __bf16 Bs[64][32];    // [d_local][k]
  const int d0 = blockIdx.x * 64;
  const int tid  = threadIdx.x;
  const int lane = tid & 31, wave = tid >> 5;
  const int half = lane >> 4, l16 = lane & 15;

  v8f acc[4];
#pragma unroll
  for (int t = 0; t < 4; ++t)
#pragma unroll
    for (int r = 0; r < 8; ++r) acc[t][r] = 0.0f;

  const int brow = tid >> 2;
  const int bcol = (tid & 3) * 8;

  for (int c0 = 0; c0 < CC; c0 += 32) {
    // B tile: async copy ewb[d0+row][c0+col..+7]
    async_to_lds_b128(&Bs[brow][bcol], ewb + (size_t)(d0 + brow) * CC + c0 + bcol);
    // A tile: xgmp fp32 -> bf16, one float4 -> one ds_store_b64 per step
#pragma unroll
    for (int i = 0; i < 4; ++i) {
      int lin4 = tid + 256 * i;      // 0..1023
      int row = lin4 >> 3;           // 0..127
      int col = (lin4 & 7) * 4;      // 0..28
      float4 v = *(const float4*)(xgmp + (size_t)row * CC + c0 + col);
      v4bf p;
      p[0] = (__bf16)v.x; p[1] = (__bf16)v.y; p[2] = (__bf16)v.z; p[3] = (__bf16)v.w;
      *(v4bf*)&As[row][col] = p;
    }
    wait_async();
    __syncthreads();
    v16bf af;
    {
      const __bf16* ap  = &As[16 * wave + l16][half * 16];
      const __bf16* ap2 = &As[16 * wave + l16][half * 16 + 8];
#pragma unroll
      for (int i = 0; i < 8; ++i) { af[i] = ap[i]; af[8 + i] = ap2[i]; }
    }
#pragma unroll
    for (int t = 0; t < 4; ++t) {
      v16bf bf;
      const __bf16* bp = &Bs[16 * t + l16][half * 16];
#pragma unroll
      for (int i = 0; i < 16; ++i) bf[i] = bp[i];
      acc[t] = __builtin_amdgcn_wmma_f32_16x16x32_bf16(false, af, false, bf,
                                                       (short)0, acc[t], false, false);
    }
    __syncthreads();
  }
#pragma unroll
  for (int t = 0; t < 4; ++t) {
    int d = d0 + 16 * t + l16;
#pragma unroll
    for (int r = 0; r < 8; ++r) {
      int bi = 16 * wave + r + 8 * half;
      xembun[(size_t)bi * DD + d] = acc[t][r] + eb[d];
    }
  }
}

// ---------------------------------------------------------------------------
// Kernel 8: classifier probs (tiny)
// ---------------------------------------------------------------------------
__global__ void __launch_bounds__(128) probs_kernel(const float* __restrict__ xembun,
                                                    const float* __restrict__ cw,
                                                    const float* __restrict__ cb,
                                                    float* __restrict__ out_probs) {
  const int b = blockIdx.x, t = threadIdx.x;
  if (t >= NCLS) return;
  const float* x = xembun + (size_t)b * DD;
  const float* w = cw + (size_t)t * DD;
  float s = cb[t];
  for (int d = 0; d < DD; ++d) s += x[d] * w[d];
  out_probs[b * NCLS + t] = s;
}

// ---------------------------------------------------------------------------
// Kernel 9: x_emb = l2n(x_emb_un)
// ---------------------------------------------------------------------------
__global__ void __launch_bounds__(32) embn_kernel(const float* __restrict__ xembun,
                                                  float* __restrict__ out_emb) {
  const int b = blockIdx.x, lane = threadIdx.x;
  const float* x = xembun + (size_t)b * DD;
  float s = 0.0f;
  for (int d = lane; d < DD; d += 32) { float v = x[d]; s += v * v; }
  for (int o = 16; o; o >>= 1) s += __shfl_xor(s, o, 32);
  float inv = 1.0f / fmaxf(sqrtf(s), 1e-12f);
  for (int d = lane; d < DD; d += 32) out_emb[(size_t)b * DD + d] = x[d] * inv;
}

// ---------------------------------------------------------------------------
extern "C" void kernel_launch(void* const* d_in, const int* in_sizes, int n_in,
                              void* d_out, int out_size, void* d_ws, size_t ws_size,
                              hipStream_t stream) {
  (void)in_sizes; (void)n_in; (void)out_size; (void)ws_size;
  const float* feat = (const float*)d_in[0];   // (B,C,H,W)
  const float* ew   = (const float*)d_in[1];   // (D,C)
  const float* eb   = (const float*)d_in[2];   // (D,)
  const float* cw   = (const float*)d_in[3];   // (NC,D)
  const float* cb   = (const float*)d_in[4];   // (NC,)
  const float* prot = (const float*)d_in[5];   // (K,D)

  float* out       = (float*)d_out;
  float* out_probs = out;                       // 128*100
  float* out_emb   = out + BB * NCLS;           // 128*512
  float* out_attr  = out + BB * NCLS + BB * DD; // 128*64

  char* ws = (char*)d_ws;
  size_t off = 0;
  auto carve = [&](size_t bytes) -> char* {
    char* p = ws + off;
    off = (off + bytes + 255) & ~(size_t)255;
    return p;
  };
  __bf16* bowT   = (__bf16*)carve((size_t)BB * NN * DD * 2); // 25.7 MB
  float*  logK   = (float*)carve((size_t)BB * KP * NN * 4);  // 6.4 MB
  float*  nrm    = (float*)carve((size_t)BB * NN * 4);
  float*  mw     = (float*)carve((size_t)BB * NN * 4);
  float*  xgmp   = (float*)carve((size_t)BB * CC * 4);
  float*  xembun = (float*)carve((size_t)BB * DD * 4);
  __bf16* pn     = (__bf16*)carve((size_t)KP * DD * 2);
  __bf16* ewb    = (__bf16*)carve((size_t)DD * CC * 2);      // 2 MB

  ew_to_bf16 <<<dim3(DD * CC / 1024), 256, 0, stream>>>(ew, ewb);
  proto_norm <<<dim3(KP),        32, 0, stream>>>(prot, pn);
  bow_gemm   <<<dim3(2, 8, BB), 256, 0, stream>>>(feat, ewb, eb, bowT);
  bow_norm   <<<dim3(NN, BB),    32, 0, stream>>>(bowT, nrm);
  cost_gemm  <<<dim3(2, 1, BB), 256, 0, stream>>>(bowT, pn, nrm, logK);
  sinkhorn   <<<dim3(BB),        32, 0, stream>>>(logK, mw, out_attr);
  gmp_pool   <<<dim3(CC / 8, BB), 256, 0, stream>>>(feat, mw, xgmp);
  emb_gemm   <<<dim3(DD / 64),  256, 0, stream>>>(xgmp, ewb, eb, xembun);
  probs_kernel<<<dim3(BB),      128, 0, stream>>>(xembun, cw, cb, out_probs);
  embn_kernel<<<dim3(BB),        32, 0, stream>>>(xembun, out_emb);
}
